// LinearRNN_45397804319088
// MI455X (gfx1250) — compile-verified, matrix operand order checked by reading, not run on hardware
//
#include <hip/hip_runtime.h>

typedef __attribute__((ext_vector_type(16))) __bf16 v16bf;
typedef __attribute__((ext_vector_type(8)))  float  v8f;

#define HID    512
#define TSEQ   4096
#define NB     16
#define NCHUNK 32
#define CHLEN  128                     // TSEQ / NCHUNK
#define MROWS  (NB * TSEQ)             // 65536 rows in the big GEMMs
#define HALF_OUT ((size_t)MROWS * HID) // 33,554,432 floats per output tensor

// ---------- bf16 split helpers (round-to-nearest-even) ----------
static __device__ __forceinline__ unsigned short f2bf_rne(float f) {
    unsigned int u = __float_as_uint(f);
    u += 0x7FFFu + ((u >> 16) & 1u);
    return (unsigned short)(u >> 16);
}
static __device__ __forceinline__ float bf2f(unsigned short h) {
    return __uint_as_float(((unsigned int)h) << 16);
}

// WMMA 16-bit fragment swizzle: element (row r in 0..511, k in 0..511) ->
// flat index in a [32 rtile][16 ktile][32 lane][16 j] bf16 array.
static __device__ __forceinline__ int swz_idx(int r, int k) {
    int kk   = k & 31;
    int half = (kk >> 3) & 1;
    int v    = ((kk >> 4) << 2) + ((kk & 7) >> 1);
    int j    = (v << 1) + (kk & 1);
    int lane = (r & 15) + (half << 4);
    return (((r >> 4) * (HID / 32) + (k >> 5)) * 32 + lane) * 16 + j;
}

// ---------- Phase 1: per-chunk (sum, double-sum) aggregates ----------
__global__ __launch_bounds__(256) void scan_phase1(const float* __restrict__ x,
                                                   float* __restrict__ agg1,
                                                   float* __restrict__ agg2) {
    int tid   = blockIdx.x * 256 + threadIdx.x;   // 16*32*512 = 262144
    int c     = tid & (HID - 1);
    int chunk = (tid >> 9) & (NCHUNK - 1);
    int b     = tid >> 14;
    size_t base = ((size_t)b * TSEQ + (size_t)chunk * CHLEN) * HID + c;
    float s1 = 0.f, s2 = 0.f;
#pragma unroll 8
    for (int j = 0; j < CHLEN; ++j) {
        float v = x[base + (size_t)j * HID];
        s1 += v; s2 += s1;
    }
    agg1[tid] = s1;
    agg2[tid] = s2;
}

// ---------- Phase 2: serial carry combine across chunks (in-place) ----------
__global__ __launch_bounds__(256) void scan_phase2(float* __restrict__ agg1,
                                                   float* __restrict__ agg2) {
    int tid = blockIdx.x * 256 + threadIdx.x;     // 16*512 = 8192 sequences
    int c = tid & (HID - 1);
    int b = tid >> 9;
    float c1 = 0.f, c2 = 0.f;
    for (int ch = 0; ch < NCHUNK; ++ch) {
        int idx = ((b * NCHUNK + ch) << 9) + c;
        float a1 = agg1[idx], a2 = agg2[idx];
        agg1[idx] = c1;                            // exclusive carries
        agg2[idx] = c2;
        c2 += (float)CHLEN * c1 + a2;
        c1 += a1;
    }
}

// ---------- Phase 3: replay with seeded carries, write Y = cumsum^2(x) ----------
__global__ __launch_bounds__(256) void scan_phase3(const float* __restrict__ x,
                                                   const float* __restrict__ agg1,
                                                   const float* __restrict__ agg2,
                                                   float* __restrict__ Y) {
    int tid   = blockIdx.x * 256 + threadIdx.x;
    int c     = tid & (HID - 1);
    int chunk = (tid >> 9) & (NCHUNK - 1);
    int b     = tid >> 14;
    size_t base = ((size_t)b * TSEQ + (size_t)chunk * CHLEN) * HID + c;
    float g1 = agg1[tid], g2 = agg2[tid];
#pragma unroll 8
    for (int j = 0; j < CHLEN; ++j) {
        g1 += x[base + (size_t)j * HID];
        g2 += g1;
        Y[base + (size_t)j * HID] = g2;
    }
}

// ---------- Wc = W2 @ W1, emitted directly in swizzled bf16 hi/lo form ----------
__global__ __launch_bounds__(256) void wc_swz_kernel(const float* __restrict__ Wih,
                                                     unsigned short* __restrict__ Whi,
                                                     unsigned short* __restrict__ Wlo) {
    int tid = blockIdx.x * 256 + threadIdx.x;     // 512*512 = 262144
    int i = tid & (HID - 1);
    int o = tid >> 9;
    const float* W1 = Wih;
    const float* W2 = Wih + HID * HID;
    float s = 0.f;
#pragma unroll 4
    for (int k = 0; k < HID; ++k) s += W2[o * HID + k] * W1[k * HID + i];
    unsigned short hi = f2bf_rne(s);
    int idx = swz_idx(o, i);                      // row = o (N), col = i (K)
    Whi[idx] = hi;
    Wlo[idx] = f2bf_rne(s - bf2f(hi));
}

// ---------- Pre-swizzle an existing (N,K) f32 weight into bf16 hi/lo ----------
__global__ __launch_bounds__(256) void wswz_kernel(const float* __restrict__ W,
                                                   unsigned short* __restrict__ Whi,
                                                   unsigned short* __restrict__ Wlo) {
    int tid = blockIdx.x * 256 + threadIdx.x;     // 262144
    int k = tid & (HID - 1);
    int n = tid >> 9;
    float f = W[(size_t)n * HID + k];
    unsigned short hi = f2bf_rne(f);
    int idx = swz_idx(n, k);
    Whi[idx] = hi;
    Wlo[idx] = f2bf_rne(f - bf2f(hi));
}

// ---------- GEMM: C[m][n] = sum_k A[m][k] * W[n][k], fp32 via bf16x3 WMMA ----------
// A: f32, converted to bf16 hi/lo + swizzled into LDS by VALU (overlaps WMMA).
// B: pre-swizzled bf16 hi/lo in global; copied with global_load_async_to_lds_b128
//    (ASYNCcnt-tracked DMA, drained with s_wait_asynccnt before the barrier).
__global__ __launch_bounds__(256) void gemm_bf16x3(const float* __restrict__ A,
                                                   const unsigned short* __restrict__ Bhi_g,
                                                   const unsigned short* __restrict__ Blo_g,
                                                   float* __restrict__ C) {
    __shared__ __align__(32) unsigned short Ahi[8 * 2 * 32 * 16];  // 16 KB
    __shared__ __align__(32) unsigned short Alo[8 * 2 * 32 * 16];  // 16 KB
    __shared__ __align__(32) unsigned short Bhi[4 * 2 * 32 * 16];  //  8 KB
    __shared__ __align__(32) unsigned short Blo[4 * 2 * 32 * 16];  //  8 KB

    const int tid  = threadIdx.x;
    const int lane = tid & 31;
    const int wave = tid >> 5;                    // 0..7 -> 16-row group
    const size_t m0 = (size_t)blockIdx.x * 128;
    const int    n0 = blockIdx.y * 64;

    const unsigned ahi_lds0 = (unsigned)(unsigned long long)(const void*)&Ahi[0]; // (touch to keep symmetry)
    const unsigned bhi_lds0 = (unsigned)(unsigned long long)(const void*)&Bhi[0];
    const unsigned blo_lds0 = (unsigned)(unsigned long long)(const void*)&Blo[0];
    (void)ahi_lds0;

    const v8f vz = {0.f, 0.f, 0.f, 0.f, 0.f, 0.f, 0.f, 0.f};
    v8f acc[4] = {vz, vz, vz, vz};

    for (int k0 = 0; k0 < HID; k0 += 64) {
        // ---- B tiles: async DMA of pre-swizzled fragment blocks, 16B per op ----
        // hi: 8 KB = 8 blocks of 1KB; block (t,kstep) is contiguous in global.
#pragma unroll
        for (int q = 0; q < 2; ++q) {
            int obyte = (tid * 2 + q) * 16;       // 0..8176, 16B granules
            int blk   = obyte >> 10;              // (t*2 + kstep)
            int t     = blk >> 1, kstep = blk & 1;
            int within = obyte & 1023;
            size_t gbase = (size_t)((((n0 >> 4) + t) * (HID / 32) + (k0 >> 5) + kstep)) * 512; // elements
            unsigned long long ghi = (unsigned long long)((const char*)Bhi_g + gbase * 2 + within);
            unsigned long long glo = (unsigned long long)((const char*)Blo_g + gbase * 2 + within);
            unsigned dhi = bhi_lds0 + (unsigned)obyte;
            unsigned dlo = blo_lds0 + (unsigned)obyte;
            asm volatile("global_load_async_to_lds_b128 %0, %1, off"
                         :: "v"(dhi), "v"(ghi) : "memory");
            asm volatile("global_load_async_to_lds_b128 %0, %1, off"
                         :: "v"(dlo), "v"(glo) : "memory");
        }

        // ---- A tile: 128 rows x 64 k, f32 -> bf16 hi/lo, swizzled into LDS ----
        for (int e = tid; e < 128 * 64; e += 256) {
            int m = e >> 6, kc = e & 63;
            const float* ap = &A[(m0 + m) * HID + k0 + kc];
            float f = *ap;
            if (k0 + 64 < HID) __builtin_prefetch(ap + 64, 0, 0); // global_prefetch_b8
            unsigned short hi = f2bf_rne(f);
            unsigned short lo = f2bf_rne(f - bf2f(hi));
            int kk = kc & 31, kstep = kc >> 5;
            int half = (kk >> 3) & 1;
            int v    = ((kk >> 4) << 2) + ((kk & 7) >> 1);
            int j    = (v << 1) + (kk & 1);
            int l    = (m & 15) + (half << 4);
            int idx  = (((m >> 4) * 2 + kstep) * 32 + l) * 16 + j;
            Ahi[idx] = hi; Alo[idx] = lo;
        }

        // Drain the async B DMA before making the tile visible block-wide.
        asm volatile("s_wait_asynccnt 0" ::: "memory");
        __syncthreads();

#pragma unroll
        for (int kstep = 0; kstep < 2; ++kstep) {
            const v16bf a_hi = *reinterpret_cast<const v16bf*>(&Ahi[((wave * 2 + kstep) * 32 + lane) * 16]);
            const v16bf a_lo = *reinterpret_cast<const v16bf*>(&Alo[((wave * 2 + kstep) * 32 + lane) * 16]);
#pragma unroll
            for (int t = 0; t < 4; ++t) {
                const v16bf b_hi = *reinterpret_cast<const v16bf*>(&Bhi[((t * 2 + kstep) * 32 + lane) * 16]);
                const v16bf b_lo = *reinterpret_cast<const v16bf*>(&Blo[((t * 2 + kstep) * 32 + lane) * 16]);
                acc[t] = __builtin_amdgcn_wmma_f32_16x16x32_bf16(false, a_hi, false, b_hi, (short)0, acc[t], false, false);
                acc[t] = __builtin_amdgcn_wmma_f32_16x16x32_bf16(false, a_hi, false, b_lo, (short)0, acc[t], false, false);
                acc[t] = __builtin_amdgcn_wmma_f32_16x16x32_bf16(false, a_lo, false, b_hi, (short)0, acc[t], false, false);
            }
        }
        __syncthreads();
    }

    // ---- write D per ISA C/D layout: VGPR r -> M = r + 8*(lane>=16), N = lane%16 ----
    const int mrow = wave * 16 + ((lane >> 4) << 3);
    const int ncol = n0 + (lane & 15);
#pragma unroll
    for (int t = 0; t < 4; ++t) {
#pragma unroll
        for (int r = 0; r < 8; ++r) {
            C[(m0 + mrow + r) * (size_t)HID + ncol + t * 16] = acc[t][r];
        }
    }
}

extern "C" void kernel_launch(void* const* d_in, const int* in_sizes, int n_in,
                              void* d_out, int out_size, void* d_ws, size_t ws_size,
                              hipStream_t stream) {
    const float* x    = (const float*)d_in[0];   // (16, 4096, 512)
    const float* Wih  = (const float*)d_in[1];   // (2, 512, 512)
    const float* Wout = (const float*)d_in[2];   // (512, 512)

    float* out     = (float*)d_out;
    float* outputs = out;                        // first tuple element
    float* hidden  = out + HALF_OUT;             // second tuple element

    float* agg1 = (float*)d_ws;                  // 262144 floats
    float* agg2 = agg1 + NB * NCHUNK * HID;      // 262144 floats
    unsigned short* wchi = (unsigned short*)(agg2 + NB * NCHUNK * HID);
    unsigned short* wclo = wchi + HID * HID;     // 4 x 512KB swizzled weights
    unsigned short* wohi = wclo + HID * HID;
    unsigned short* wolo = wohi + HID * HID;     // total ws: 4 MB

    // Y = cumsum^2_t(x), staged in the `outputs` slot (overwritten by GEMM2).
    scan_phase1<<<1024, 256, 0, stream>>>(x, agg1, agg2);
    scan_phase2<<<32,   256, 0, stream>>>(agg1, agg2);
    scan_phase3<<<1024, 256, 0, stream>>>(x, agg1, agg2, outputs);

    // Wc = W2 @ W1 and Wout, pre-swizzled to bf16 hi/lo fragment layout.
    wc_swz_kernel<<<1024, 256, 0, stream>>>(Wih, wchi, wclo);
    wswz_kernel<<<1024, 256, 0, stream>>>(Wout, wohi, wolo);

    dim3 grid(MROWS / 128, HID / 64);            // (512, 8)
    // hidden = Y @ Wc^T
    gemm_bf16x3<<<grid, 256, 0, stream>>>(outputs, wchi, wclo, hidden);
    // outputs = hidden @ Wout^T
    gemm_bf16x3<<<grid, 256, 0, stream>>>(hidden, wohi, wolo, outputs);
}
